// PNN_46127948759431
// MI455X (gfx1250) — compile-verified, matrix operand order checked by reading, not run on hardware
//
#include <hip/hip_runtime.h>
#include <hip/hip_bf16.h>
#include <math.h>

typedef _Float16 v8h  __attribute__((ext_vector_type(8)));
typedef _Float16 v16h __attribute__((ext_vector_type(16)));
typedef float    v8f  __attribute__((ext_vector_type(8)));

#define WAVES_PER_WG 4
#define TPB (WAVES_PER_WG * 32)
#define ROWS 16

// Statically-typed LDS: keeps addrspace(3) provenance so all fragment traffic
// lowers to ds_load_b128/ds_store_b16 (no generic-pointer null checks, no flat).
struct Smem {
  float cw1f[256];              // [0..124]=cw1 row0, [125..249]=row1
  float cb1f[128];
  float cb2f[128];
  float cb3f[16];
  float hw1f[256];              // hw1 (4,64) row-major f32
  float hb1f[64], hb2f[64], hb3f[64], hb4f[64], hb5f[64], hw6f[64];
  float Sf[16];
  float waveF[WAVES_PER_WG][288];  // per wave: x 64 | theta 64 | dH 64 | phi 64 | cpl 32
  _Float16 cw2T[128*128];       // [n][k] = cw2[k][n], zero padded
  _Float16 cw3T[16*128];        // [n][k] = cw3[k][n], rows 2..15 zero
  _Float16 hwT[4][4096];        // hw2T..hw5T: [n][k] = W[k][n]  (forward B)
  _Float16 hwRM[4][4096];       // hw2..hw5 row-major            (backward B)
  _Float16 hw1b[1024];          // hw1 row-major padded to 16 rows
  _Float16 act[WAVES_PER_WG][2048];   // coupling L1 out / backward g0|g1 ping-pong
  _Float16 ab[WAVES_PER_WG][5*1024];  // a1..a5 (stride 64); also aliases coupling act2
};
// ~171 KB < 320 KB gfx1250 LDS

// tanh: hardware V_TANH_F32 if the builtin exists, else branch-free exp2/rcp.
__device__ __forceinline__ float fast_tanh(float x) {
#if __has_builtin(__builtin_amdgcn_tanhf)
  return __builtin_amdgcn_tanhf(x);
#else
  float e = __builtin_amdgcn_exp2f(x * 2.8853900817779268f);
  return 1.0f - 2.0f * __builtin_amdgcn_rcpf(e + 1.0f);
#endif
}

// Load one 16-half A/B fragment: halves [0..7] and [16..23] from p
// (per-lane layout of V_WMMA_*_16X16X32_F16; kblock/hi offsets applied by caller).
__device__ __forceinline__ v16h ld_frag(const _Float16* p) {
  v8h lo = *(const v8h*)(p);
  v8h hi = *(const v8h*)(p + 16);
  return __builtin_shufflevector(lo, hi, 0,1,2,3,4,5,6,7,8,9,10,11,12,13,14,15);
}

__device__ __forceinline__ v8f wmma16(v16h a, v16h b, v8f c) {
  return __builtin_amdgcn_wmma_f32_16x16x32_f16(false, a, false, b, (short)0, c, false, false);
}

// Forward layer: out[16xN] = act( A[16xK] @ W[KxN] + bias ), W stored transposed
// as BT[N][K] (row n = column n of W). Strides in halves. ACT: 0=relu, 1=tanh.
// A-fragments are preloaded once and reused across all N/16 tiles.
template<int K, int N, int ACT>
__device__ __forceinline__ void fwd_layer(const _Float16* A, int aStrideH,
                                          const _Float16* BT, int bStrideH,
                                          const float* bias,
                                          _Float16* out, int outStrideH, int lane)
{
  const int l15 = lane & 15;
  const int hi  = lane >> 4;
  v16h afrag[K/32];
  #pragma unroll
  for (int k = 0; k < K/32; ++k)
    afrag[k] = ld_frag(A + l15*aStrideH + k*32 + hi*8);
  #pragma unroll
  for (int t = 0; t < N/16; ++t) {
    const int n = t*16 + l15;
    float bv = bias[n];
    v8f c = { bv,bv,bv,bv,bv,bv,bv,bv };
    #pragma unroll
    for (int k = 0; k < K/32; ++k) {
      v16h b = ld_frag(BT + n*bStrideH + k*32 + hi*8);
      c = wmma16(afrag[k], b, c);
    }
    #pragma unroll
    for (int j = 0; j < 8; ++j) {
      float v = c[j];
      if (ACT == 0) v = v > 0.0f ? v : 0.0f;
      else          v = fast_tanh(v);
      out[(hi*8 + j)*outStrideH + n] = (_Float16)v;
    }
  }
  __syncthreads();
}

// Backward layer: out = (G[16x64] @ W^T) .* deriv(Aact). For g@W^T the WMMA
// B-matrix is W^T, whose column n is row n of W -> Wrm is W row-major (64x64).
// DER: 0=relu', 1=tanh'. G-fragments preloaded once.
template<int DER>
__device__ __forceinline__ void bwd_layer(const _Float16* G, const _Float16* Wrm,
                                          const _Float16* Aact, _Float16* out, int lane)
{
  const int l15 = lane & 15;
  const int hi  = lane >> 4;
  v16h gfrag[2];
  #pragma unroll
  for (int k = 0; k < 2; ++k)
    gfrag[k] = ld_frag(G + l15*64 + k*32 + hi*8);
  #pragma unroll
  for (int t = 0; t < 4; ++t) {
    const int n = t*16 + l15;
    v8f c = { 0,0,0,0,0,0,0,0 };
    #pragma unroll
    for (int k = 0; k < 2; ++k) {
      v16h b = ld_frag(Wrm + n*64 + k*32 + hi*8);
      c = wmma16(gfrag[k], b, c);
    }
    #pragma unroll
    for (int j = 0; j < 8; ++j) {
      const int m = hi*8 + j;
      float v = c[j];
      float a = (float)Aact[m*64 + n];
      if (DER == 0) v = (a > 0.0f) ? v : 0.0f;  // relu'
      else          v = v * (1.0f - a*a);       // tanh'
      out[m*64 + n] = (_Float16)v;
    }
  }
  __syncthreads();
}

// Coupling net 2->125->125->2 on 16 rows. in[m*inStride + {0,1}], out2[m*2+{0,1}].
// act2 may alias the (dead at call time) a1..a5 region.
__device__ __forceinline__ void coupling_net(const float* in, int inStride, float* out2,
    const float* cw1f, const float* cb1f, const float* cb2f, const float* cb3f,
    const _Float16* cw2T, const _Float16* cw3T,
    _Float16* act, _Float16* act2, int lane)
{
  // L1 (K=2): VALU, relu, padded to 128 cols
  for (int idx = lane; idx < 16*128; idx += 32) {
    int m = idx >> 7, n = idx & 127;
    float v = 0.0f;
    if (n < 125) {
      v = in[m*inStride + 0]*cw1f[n] + in[m*inStride + 1]*cw1f[125 + n] + cb1f[n];
      v = v > 0.0f ? v : 0.0f;
    }
    act[m*128 + n] = (_Float16)v;
  }
  __syncthreads();
  // L2 (125x125 padded to 128x128), tanh
  fwd_layer<128, 128, 1>(act, 128, cw2T, 128, cb2f, act2, 128, lane);
  // L3 (125->2): K=128, single N tile, cols 0..1 valid
  {
    const int l15 = lane & 15, hi = lane >> 4;
    float bv = (l15 < 2) ? cb3f[l15] : 0.0f;
    v8f c = { bv,bv,bv,bv,bv,bv,bv,bv };
    #pragma unroll
    for (int k = 0; k < 4; ++k) {
      v16h a = ld_frag(act2 + l15*128 + k*32 + hi*8);
      v16h b = ld_frag(cw3T + l15*128 + k*32 + hi*8);
      c = wmma16(a, b, c);
    }
    if (l15 < 2) {
      #pragma unroll
      for (int j = 0; j < 8; ++j) out2[(hi*8 + j)*2 + l15] = c[j];
    }
    __syncthreads();
  }
}

__global__ __launch_bounds__(TPB)
void phnn_kernel(const float* __restrict__ x,
                 const float* __restrict__ cw1, const float* __restrict__ cb1,
                 const float* __restrict__ cw2, const float* __restrict__ cb2,
                 const float* __restrict__ cw3, const float* __restrict__ cb3,
                 const float* __restrict__ hw1, const float* __restrict__ hb1,
                 const float* __restrict__ hw2, const float* __restrict__ hb2,
                 const float* __restrict__ hw3, const float* __restrict__ hb3,
                 const float* __restrict__ hw4, const float* __restrict__ hb4,
                 const float* __restrict__ hw5, const float* __restrict__ hb5,
                 const float* __restrict__ hw6,
                 const float* __restrict__ Smat,
                 const float* __restrict__ dtq_p, const float* __restrict__ dtp_p,
                 const float* __restrict__ alpha_p,
                 float* __restrict__ out, int rows)
{
  __shared__ Smem sm;

  // ---- cooperative weight load (f32 -> f16, with transposes/padding) ----
  const int tid = threadIdx.x;
  for (int i = tid; i < 256; i += TPB) sm.cw1f[i] = (i < 250) ? cw1[i] : 0.0f;
  for (int i = tid; i < 128; i += TPB) sm.cb1f[i] = (i < 125) ? cb1[i] : 0.0f;
  for (int i = tid; i < 128; i += TPB) sm.cb2f[i] = (i < 125) ? cb2[i] : 0.0f;
  for (int i = tid; i < 16;  i += TPB) sm.cb3f[i] = (i < 2)   ? cb3[i] : 0.0f;
  for (int i = tid; i < 256; i += TPB) sm.hw1f[i] = hw1[i];
  for (int i = tid; i < 64;  i += TPB) {
    sm.hb1f[i] = hb1[i]; sm.hb2f[i] = hb2[i]; sm.hb3f[i] = hb3[i];
    sm.hb4f[i] = hb4[i]; sm.hb5f[i] = hb5[i]; sm.hw6f[i] = hw6[i];
  }
  for (int i = tid; i < 16; i += TPB) sm.Sf[i] = Smat[i];
  for (int i = tid; i < 128*128; i += TPB) {
    int n = i >> 7, k = i & 127;
    sm.cw2T[i] = (n < 125 && k < 125) ? (_Float16)cw2[k*125 + n] : (_Float16)0.0f;
  }
  for (int i = tid; i < 16*128; i += TPB) {
    int n = i >> 7, k = i & 127;
    sm.cw3T[i] = (n < 2 && k < 125) ? (_Float16)cw3[k*2 + n] : (_Float16)0.0f;
  }
  {
    const float* hws[4] = { hw2, hw3, hw4, hw5 };
    #pragma unroll
    for (int w = 0; w < 4; ++w) {
      const float* W = hws[w];
      for (int i = tid; i < 4096; i += TPB) {
        int n = i >> 6, k = i & 63;
        sm.hwT[w][i]  = (_Float16)W[k*64 + n];
        sm.hwRM[w][i] = (_Float16)W[i];
      }
    }
  }
  for (int i = tid; i < 1024; i += TPB) {
    int n = i >> 6, k = i & 63;
    sm.hw1b[i] = (n < 4) ? (_Float16)hw1[n*64 + k] : (_Float16)0.0f;
  }
  __syncthreads();

  // ---- per-wave batch tile ----
  const int lane = tid & 31;
  const int wv   = tid >> 5;
  int tr = (blockIdx.x * WAVES_PER_WG + wv) * ROWS;
  if (tr > rows - ROWS) tr = rows - ROWS;   // keep barriers uniform at the tail

  float* xt   = sm.waveF[wv];
  float* thf  = xt  + 64;   // theta [16][4]
  float* dHf  = thf + 64;   // dH    [16][4]
  float* phf  = dHf + 64;   // phi   [16][4]
  float* cplf = phf + 64;   // coupling output [16][2]

  _Float16* act  = sm.act[wv];
  _Float16* ab   = sm.ab[wv];     // a1..a5 at ab + L*1024, stride 64
  _Float16* act2 = ab;            // coupling L2 staging aliases dead a1..a2 region
  _Float16* g0   = act;           // backward ping-pong (act region reused)
  _Float16* g1   = act + 1024;

  // load x tile (16 rows x 4 f32)
  for (int i = lane; i < 64; i += 32) xt[i] = x[tr*4 + i];
  __syncthreads();

  // coupling #1 on x1
  coupling_net(xt, 4, cplf, sm.cw1f, sm.cb1f, sm.cb2f, sm.cb3f,
               sm.cw2T, sm.cw3T, act, act2, lane);

  // theta = [x1, x2 + coupling(x1)]
  for (int i = lane; i < 64; i += 32) {
    int m = i >> 2, k = i & 3;
    float v = xt[i];
    if (k >= 2) v += cplf[m*2 + (k - 2)];
    thf[i] = v;
  }
  __syncthreads();

  // H-net L1 (K=4): VALU -> a1
  for (int i = lane; i < 1024; i += 32) {
    int m = i >> 6, n = i & 63;
    float v = sm.hb1f[n];
    v += thf[m*4 + 0]*sm.hw1f[0*64 + n];
    v += thf[m*4 + 1]*sm.hw1f[1*64 + n];
    v += thf[m*4 + 2]*sm.hw1f[2*64 + n];
    v += thf[m*4 + 3]*sm.hw1f[3*64 + n];
    v = v > 0.0f ? v : 0.0f;
    ab[i] = (_Float16)v;
  }
  __syncthreads();

  // H-net L2..L5 (WMMA)
  fwd_layer<64, 64, 0>(ab + 0*1024, 64, sm.hwT[0], 64, sm.hb2f, ab + 1*1024, 64, lane); // relu
  fwd_layer<64, 64, 1>(ab + 1*1024, 64, sm.hwT[1], 64, sm.hb3f, ab + 2*1024, 64, lane); // tanh
  fwd_layer<64, 64, 1>(ab + 2*1024, 64, sm.hwT[2], 64, sm.hb4f, ab + 3*1024, 64, lane); // tanh
  fwd_layer<64, 64, 1>(ab + 3*1024, 64, sm.hwT[3], 64, sm.hb5f, ab + 4*1024, 64, lane); // tanh

  // backward seed: d_pre5 = hw6 .* (1 - a5^2)
  for (int i = lane; i < 1024; i += 32) {
    int n = i & 63;
    float a = (float)ab[4*1024 + i];
    g0[i] = (_Float16)(sm.hw6f[n] * (1.0f - a*a));
  }
  __syncthreads();

  // backward chain (WMMA): g @ W^T, then elementwise deriv of previous act
  bwd_layer<1>(g0, sm.hwRM[3], ab + 3*1024, g1, lane); // via hw5 -> d_pre4
  bwd_layer<1>(g1, sm.hwRM[2], ab + 2*1024, g0, lane); // via hw4 -> d_pre3
  bwd_layer<0>(g0, sm.hwRM[1], ab + 1*1024, g1, lane); // via hw3 -> d_pre2
  bwd_layer<0>(g1, sm.hwRM[0], ab + 0*1024, g0, lane); // via hw2 -> d_pre1

  // dH = d_pre1 @ hw1^T  (K=64, one N tile, cols 0..3 valid)
  {
    const int l15 = lane & 15, hi = lane >> 4;
    v8f c = { 0,0,0,0,0,0,0,0 };
    #pragma unroll
    for (int k = 0; k < 2; ++k) {
      v16h a = ld_frag(g0      + l15*64 + k*32 + hi*8);
      v16h b = ld_frag(sm.hw1b + l15*64 + k*32 + hi*8);
      c = wmma16(a, b, c);
    }
    if (l15 < 4) {
      #pragma unroll
      for (int j = 0; j < 8; ++j) dHf[(hi*8 + j)*4 + l15] = c[j];
    }
    __syncthreads();
  }

  // phi = theta + DT * [dz1, dz2]  (symplectic + skew term, VALU)
  const float dtq   = dtq_p[0];
  const float dtp   = dtp_p[0];
  const float alpha = alpha_p[0];
  for (int i = lane; i < 64; i += 32) {
    int m = i >> 2, k = i & 3;
    float tv[4] = { thf[m*4+0], thf[m*4+1], thf[m*4+2], thf[m*4+3] };
    float dz;
    if (k < 2) {
      // dz1_i = dH[:,2+i]*dt_q + alpha * sum_j theta_j * (S[i][j]-S[j][i])
      float s = 0.0f;
      #pragma unroll
      for (int j = 0; j < 4; ++j) s += tv[j] * (sm.Sf[k*4 + j] - sm.Sf[j*4 + k]);
      dz = dHf[m*4 + 2 + k]*dtq + alpha*s;
    } else {
      // dz2_i = -dH[:,i]*dt_p + alpha * sum_j theta_j * (S[j][c]-S[c][j]), c = k
      float s = 0.0f;
      #pragma unroll
      for (int j = 0; j < 4; ++j) s += tv[j] * (sm.Sf[j*4 + k] - sm.Sf[k*4 + j]);
      dz = -dHf[m*4 + (k - 2)]*dtp + alpha*s;
    }
    phf[i] = tv[k] + 0.1f*dz;
  }
  __syncthreads();

  // coupling #2 on y1 = phi[:, :2]
  coupling_net(phf, 4, cplf, sm.cw1f, sm.cb1f, sm.cb2f, sm.cb3f,
               sm.cw2T, sm.cw3T, act, act2, lane);

  // output: [y1, y2 - coupling(y1)]
  for (int i = lane; i < 64; i += 32) {
    int m = i >> 2, k = i & 3;
    float v = phf[i];
    if (k >= 2) v -= cplf[m*2 + (k - 2)];
    out[tr*4 + i] = v;
  }
}

extern "C" void kernel_launch(void* const* d_in, const int* in_sizes, int n_in,
                              void* d_out, int out_size, void* d_ws, size_t ws_size,
                              hipStream_t stream)
{
  const float* x   = (const float*)d_in[0];
  const float* cw1 = (const float*)d_in[1];
  const float* cb1 = (const float*)d_in[2];
  const float* cw2 = (const float*)d_in[3];
  const float* cb2 = (const float*)d_in[4];
  const float* cw3 = (const float*)d_in[5];
  const float* cb3 = (const float*)d_in[6];
  const float* hw1 = (const float*)d_in[7];
  const float* hb1 = (const float*)d_in[8];
  const float* hw2 = (const float*)d_in[9];
  const float* hb2 = (const float*)d_in[10];
  const float* hw3 = (const float*)d_in[11];
  const float* hb3 = (const float*)d_in[12];
  const float* hw4 = (const float*)d_in[13];
  const float* hb4 = (const float*)d_in[14];
  const float* hw5 = (const float*)d_in[15];
  const float* hb5 = (const float*)d_in[16];
  const float* hw6 = (const float*)d_in[17];
  // d_in[18] = hb6: constant offset, vanishes in grad -> unused
  const float* S   = (const float*)d_in[19];
  const float* dtq = (const float*)d_in[20];
  const float* dtp = (const float*)d_in[21];
  const float* alp = (const float*)d_in[22];
  float* out = (float*)d_out;

  const int rows = in_sizes[0] / 4;
  const int blocks = (rows + WAVES_PER_WG*ROWS - 1) / (WAVES_PER_WG*ROWS);

  phnn_kernel<<<blocks, TPB, 0, stream>>>(
      x, cw1, cb1, cw2, cb2, cw3, cb3,
      hw1, hb1, hw2, hb2, hw3, hb3, hw4, hb4, hw5, hb5, hw6,
      S, dtq, dtp, alp, out, rows);
}
